// RTNLinear_81750407512101
// MI455X (gfx1250) — matrix operand order repaired and checked by reading, not compiled, not run on hardware
//
#include <hip/hip_runtime.h>

// CDNA5 (gfx1250) wave32 quantized-linear:
//   y[m,n] = scale[n] * sum_k x[m,k] * wq[n,k] + bias[n]
//
// Fast path (ws_size >= 96 MiB):
//   pass 1: x f32 -> f16, wq int32 -> f16 (exact for [0,127]) into d_ws
//   pass 2: f16 GEMM, block tile 256x128, wave tile 64x64; LDS tiles filled by
//           TENSOR_LOAD_TO_LDS (TDM) with descriptor padding producing the
//           bank-conflict-free 72-half row stride; hot loop is purely
//           ds_load_b128 + v_wmma_f32_16x16x32_f16 (1 LDS read per WMMA).
// Fallback: fused convert-in-loop GEMM.

typedef __attribute__((ext_vector_type(16))) _Float16 v16h;
typedef __attribute__((ext_vector_type(8)))  _Float16 v8h;
typedef __attribute__((ext_vector_type(4)))  _Float16 v4h;
typedef __attribute__((ext_vector_type(8)))  float    v8f;
typedef __attribute__((ext_vector_type(4)))  unsigned int v4u;
typedef __attribute__((ext_vector_type(8)))  int      v8i;
typedef __attribute__((ext_vector_type(4)))  int      v4i;

#define TILE_K 64
#define LDS_STRIDE 72  // 64 + 8 pad: 144B rows (16B aligned), 36-dword stride -> conflict-free b128

// GEMM (TDM path) block tile
#define GM_TILE_M 256
#define GM_TILE_N 128
// Fallback block tile
#define FB_TILE_M 128
#define FB_TILE_N 128

// ---------------------------------------------------------------------------
// TDM: one 2D tile load (rows x 64 f16, global row stride = stride_elems) into
// LDS at lds_off; 16B LDS padding after each 128B row -> 144B LDS row stride.
// D# layout per CDNA5 ISA.
// ---------------------------------------------------------------------------
__device__ __forceinline__ void tdm_load_tile(unsigned lds_off, const void* gptr,
                                              unsigned rows, unsigned stride_elems) {
    unsigned long long ga = (unsigned long long)(__SIZE_TYPE__)gptr;
    v4u g0;
    g0[0] = 1u;                                        // count=1 (valid), user mode
    g0[1] = lds_off;                                   // lds_addr (bytes)
    g0[2] = (unsigned)(ga & 0xFFFFFFFFu);              // global_addr[31:0]
    g0[3] = (unsigned)((ga >> 32) & 0x1FFFFFFu)        // global_addr[56:32]
          | (2u << 30);                                // type = 2 ("image")
    const unsigned td0 = stride_elems;                 // tensor_dim0 = K
    const unsigned td1 = 0x40000000u;                  // huge tensor_dim1 (no OOB clip)
    v8i g1;
    g1[0] = (int)((1u << 16)        // data_size = 1 -> 2 bytes (f16)
                | (1u << 20)        // pad_enable
                | (4u << 22)        // pad_interval = 32 dwords (128B) between pads
                | (3u << 25));      // pad_amount   = 4 dwords (16B)
    g1[1] = (int)((td0 & 0xFFFFu) << 16);                             // tensor_dim0[15:0]
    g1[2] = (int)(((td0 >> 16) & 0xFFFFu) | ((td1 & 0xFFFFu) << 16)); // dim0[31:16]|dim1[15:0]
    g1[3] = (int)(((td1 >> 16) & 0xFFFFu) | (64u << 16));             // dim1[31:16]|tile_dim0=64
    g1[4] = (int)rows;                                                // tile_dim1, tile_dim2=0
    g1[5] = (int)td0;                                                 // tensor_dim0_stride[31:0]
    g1[6] = 0;
    g1[7] = 0;
    v4i z = (v4i)0;
#if __clang_major__ >= 23
    __builtin_amdgcn_tensor_load_to_lds(g0, g1, z, z, (v8i)0, 0);
#else
    __builtin_amdgcn_tensor_load_to_lds(g0, g1, z, z, 0);
#endif
}

// ---------------------------------------------------------------------------
// Pass 1: dtype conversion kernels (memory bound, ~12us total at 23.3 TB/s)
// ---------------------------------------------------------------------------
__global__ __launch_bounds__(256) void cvt_f32_to_f16(const float4* __restrict__ in,
                                                      v4h* __restrict__ out, int n4) {
    int i = blockIdx.x * blockDim.x + threadIdx.x;
    if (i < n4) {
        float4 f = in[i];
        v4h h;
        h[0] = (_Float16)f.x; h[1] = (_Float16)f.y;
        h[2] = (_Float16)f.z; h[3] = (_Float16)f.w;
        out[i] = h;
    }
}

__global__ __launch_bounds__(256) void cvt_i32_to_f16(const int4* __restrict__ in,
                                                      v4h* __restrict__ out, int n4) {
    int i = blockIdx.x * blockDim.x + threadIdx.x;
    if (i < n4) {
        int4 w = in[i];
        v4h h;                              // exact: values in [-2048, 2048]
        h[0] = (_Float16)(float)w.x; h[1] = (_Float16)(float)w.y;
        h[2] = (_Float16)(float)w.z; h[3] = (_Float16)(float)w.w;
        out[i] = h;
    }
}

// ---------------------------------------------------------------------------
// Pass 2: f16 GEMM, TDM-fed LDS double buffer, 64x64 wave tile WMMA compute
// ---------------------------------------------------------------------------
__global__ __launch_bounds__(256) void gemm_f16_tdm(
    const _Float16* __restrict__ xh,   // [M][K]
    const _Float16* __restrict__ wh,   // [N][K]
    const float* __restrict__ scale, const float* __restrict__ bias,
    float* __restrict__ out, int M, int N, int K)
{
    __shared__ _Float16 As[2][GM_TILE_M][LDS_STRIDE];   // 2*256*72*2B = 72KB
    __shared__ _Float16 Bs[2][GM_TILE_N][LDS_STRIDE];   // 2*128*72*2B = 36KB

    const int tid  = threadIdx.x;
    const int lane = tid & 31;
    const int wave = tid >> 5;
    const int wm   = wave & 3;   // 0..3 : 64-row slab of 256-row block tile
    const int wn   = wave >> 2;  // 0..1 : 64-col slab of 128-col block tile
    const int blockN = blockIdx.x * GM_TILE_N;
    const int blockM = blockIdx.y * GM_TILE_M;

    v8f acc[4][4] = {};          // 16 fragments x 8 VGPRs = 128 VGPRs of f32 accum
    const int NT = K / TILE_K;

    auto tdm_issue = [&](int buf, int kt) {
        tdm_load_tile((unsigned)(__SIZE_TYPE__)&As[buf][0][0],
                      xh + (size_t)blockM * K + (size_t)kt * TILE_K, GM_TILE_M, K);
        tdm_load_tile((unsigned)(__SIZE_TYPE__)&Bs[buf][0][0],
                      wh + (size_t)blockN * K + (size_t)kt * TILE_K, GM_TILE_N, K);
    };

    auto compute = [&](int buf) {
#pragma unroll
        for (int ks = 0; ks < 2; ++ks) {
            // A fragment (16-bit A 16x32 layout, ISA §7.12.2):
            // lane<16: M=lane, K {0..7},{16..23}; lane>=16: M=lane-16, K {8..15},{24..31}
            const int arow = wm * 64 + (lane & 15);
            const int akb  = ks * 32 + ((lane & 16) ? 8 : 0);
            v16h afrag[4];
#pragma unroll
            for (int mf = 0; mf < 4; ++mf) {
                const _Float16* p = &As[buf][arow + mf * 16][akb];
                union { v16h v; v8h h[2]; } u;
                u.h[0] = *(const v8h*)(p);        // ds_load_b128
                u.h[1] = *(const v8h*)(p + 16);   // ds_load_b128
                afrag[mf] = u.v;
            }
            // B fragment (32x16 K x N): lane holds N=lane%16, K {0..7},{16..23} / {8..15},{24..31}
            const int brow = wn * 64 + (lane & 15);
            const int bkb  = ks * 32 + ((lane & 16) ? 8 : 0);
            v16h bfrag[4];
#pragma unroll
            for (int nf = 0; nf < 4; ++nf) {
                const _Float16* p = &Bs[buf][brow + nf * 16][bkb];
                union { v16h v; v8h h[2]; } u;
                u.h[0] = *(const v8h*)(p);
                u.h[1] = *(const v8h*)(p + 16);
                bfrag[nf] = u.v;
            }
#pragma unroll
            for (int mf = 0; mf < 4; ++mf)
#pragma unroll
                for (int nf = 0; nf < 4; ++nf)
                    acc[mf][nf] = __builtin_amdgcn_wmma_f32_16x16x32_f16(
                        false, afrag[mf], false, bfrag[nf],
                        (short)0, acc[mf][nf], false, false);
        }
    };

    // Prologue: wave 0 DMAs tile 0 into buffer 0.
    if (wave == 0) tdm_issue(0, 0);

    for (int kt = 0; kt < NT; ++kt) {
        const int buf = kt & 1;
        if (wave == 0) {
            if (kt + 1 < NT) {
                tdm_issue(buf ^ 1, kt + 1);                 // prefetch next tile
                __builtin_amdgcn_s_wait_tensorcnt(2);       // tile kt done (in-order)
            } else {
                __builtin_amdgcn_s_wait_tensorcnt(0);
            }
        }
        __syncthreads();          // tile kt visible to all waves
        compute(buf);
        __syncthreads();          // all reads of buf done before it is overwritten
    }

    // Epilogue: fused dequant-scale + bias. Lane owns one column n per fragment.
#pragma unroll
    for (int nf = 0; nf < 4; ++nf) {
        const int n = blockN + wn * 64 + nf * 16 + (lane & 15);
        const float s = scale[n];
        const float b = bias[n];
        const int mbase = blockM + wm * 64 + ((lane & 16) ? 8 : 0);
#pragma unroll
        for (int mf = 0; mf < 4; ++mf)
#pragma unroll
            for (int v = 0; v < 8; ++v)
                out[(size_t)(mbase + mf * 16 + v) * N + n] = acc[mf][nf][v] * s + b;
    }
}

// ---------------------------------------------------------------------------
// Fallback: fused convert-in-loop GEMM (used when d_ws is too small)
// ---------------------------------------------------------------------------
__global__ __launch_bounds__(256) void rtn_linear_wmma(
    const float* __restrict__ x, const int* __restrict__ wq,
    const float* __restrict__ scale, const float* __restrict__ bias,
    float* __restrict__ out, int M, int N, int K)
{
    __shared__ _Float16 As[2][FB_TILE_M][LDS_STRIDE];
    __shared__ _Float16 Bs[2][FB_TILE_N][LDS_STRIDE];

    const int tid  = threadIdx.x;
    const int lane = tid & 31;
    const int wave = tid >> 5;
    const int wm   = wave & 1;
    const int wn   = wave >> 1;
    const int blockN = blockIdx.x * FB_TILE_N;
    const int blockM = blockIdx.y * FB_TILE_M;
    const int lr = tid >> 4;
    const int lk = (tid & 15) * 4;

    float4 aReg[8];
    int4   bReg[8];
    v8f acc[4][2] = {};
    const int NT = K / TILE_K;

    auto loadRegs = [&](int kt) {
        const int kBase = kt * TILE_K;
#pragma unroll
        for (int it = 0; it < 8; ++it) {
            const int r = lr + it * 16;
            aReg[it] = *(const float4*)(x  + (size_t)(blockM + r) * K + kBase + lk);
            bReg[it] = *(const int4*)  (wq + (size_t)(blockN + r) * K + kBase + lk);
        }
    };
    auto storeLds = [&](int buf) {
#pragma unroll
        for (int it = 0; it < 8; ++it) {
            const int r = lr + it * 16;
            v4h a, b;
            a[0] = (_Float16)aReg[it].x;  a[1] = (_Float16)aReg[it].y;
            a[2] = (_Float16)aReg[it].z;  a[3] = (_Float16)aReg[it].w;
            b[0] = (_Float16)(float)bReg[it].x;  b[1] = (_Float16)(float)bReg[it].y;
            b[2] = (_Float16)(float)bReg[it].z;  b[3] = (_Float16)(float)bReg[it].w;
            *(v4h*)&As[buf][r][lk] = a;
            *(v4h*)&Bs[buf][r][lk] = b;
        }
    };
    auto compute = [&](int buf) {
#pragma unroll
        for (int ks = 0; ks < 2; ++ks) {
            const int arow = wm * 64 + (lane & 15);
            const int akb  = ks * 32 + ((lane & 16) ? 8 : 0);
            v16h afrag[4];
#pragma unroll
            for (int mf = 0; mf < 4; ++mf) {
                const _Float16* p = &As[buf][arow + mf * 16][akb];
                union { v16h v; v8h h[2]; } u;
                u.h[0] = *(const v8h*)(p);
                u.h[1] = *(const v8h*)(p + 16);
                afrag[mf] = u.v;
            }
            const int brow = wn * 32 + (lane & 15);
            const int bkb  = ks * 32 + ((lane & 16) ? 8 : 0);
            v16h bfrag[2];
#pragma unroll
            for (int nf = 0; nf < 2; ++nf) {
                const _Float16* p = &Bs[buf][brow + nf * 16][bkb];
                union { v16h v; v8h h[2]; } u;
                u.h[0] = *(const v8h*)(p);
                u.h[1] = *(const v8h*)(p + 16);
                bfrag[nf] = u.v;
            }
#pragma unroll
            for (int mf = 0; mf < 4; ++mf)
#pragma unroll
                for (int nf = 0; nf < 2; ++nf)
                    acc[mf][nf] = __builtin_amdgcn_wmma_f32_16x16x32_f16(
                        false, afrag[mf], false, bfrag[nf],
                        (short)0, acc[mf][nf], false, false);
        }
    };

    loadRegs(0);
    storeLds(0);
    __syncthreads();
    for (int kt = 0; kt < NT; ++kt) {
        const int buf = kt & 1;
        if (kt + 1 < NT) loadRegs(kt + 1);
        compute(buf);
        if (kt + 1 < NT) storeLds(buf ^ 1);
        __syncthreads();
    }
#pragma unroll
    for (int nf = 0; nf < 2; ++nf) {
        const int n = blockN + wn * 32 + nf * 16 + (lane & 15);
        const float s = scale[n];
        const float b = bias[n];
        const int mbase = blockM + wm * 64 + ((lane & 16) ? 8 : 0);
#pragma unroll
        for (int mf = 0; mf < 4; ++mf)
#pragma unroll
            for (int v = 0; v < 8; ++v)
                out[(size_t)(mbase + mf * 16 + v) * N + n] = acc[mf][nf][v] * s + b;
    }
}

extern "C" void kernel_launch(void* const* d_in, const int* in_sizes, int n_in,
                              void* d_out, int out_size, void* d_ws, size_t ws_size,
                              hipStream_t stream) {
    const float* x     = (const float*)d_in[0];
    const int*   wq    = (const int*)  d_in[1];
    const float* scale = (const float*)d_in[2];
    const float* bias  = (const float*)d_in[3];
    float*       out   = (float*)d_out;

    const int N = in_sizes[2];            // D_OUT
    const int K = in_sizes[1] / N;        // D_IN
    const int M = in_sizes[0] / K;        // B*S

    const size_t xh_bytes = (size_t)M * K * sizeof(_Float16);
    const size_t wh_bytes = (size_t)N * K * sizeof(_Float16);

    if (ws_size >= xh_bytes + wh_bytes && (M % GM_TILE_M) == 0) {
        _Float16* xh = (_Float16*)d_ws;
        _Float16* wh = (_Float16*)((char*)d_ws + xh_bytes);

        const int nx4 = (M * K) / 4;
        const int nw4 = (N * K) / 4;
        cvt_f32_to_f16<<<(nx4 + 255) / 256, 256, 0, stream>>>(
            (const float4*)x, (v4h*)xh, nx4);
        cvt_i32_to_f16<<<(nw4 + 255) / 256, 256, 0, stream>>>(
            (const int4*)wq, (v4h*)wh, nw4);

        dim3 grid(N / GM_TILE_N, M / GM_TILE_M);
        gemm_f16_tdm<<<grid, 256, 0, stream>>>(xh, wh, scale, bias, out, M, N, K);
    } else {
        dim3 grid(N / FB_TILE_N, M / FB_TILE_M);
        rtn_linear_wmma<<<grid, 256, 0, stream>>>(x, wq, scale, bias, out, M, N, K);
    }
}
